// CrossModalAttention_80762565034528
// MI455X (gfx1250) — matrix operand order repaired and checked by reading, not compiled, not run on hardware
//
#include <hip/hip_runtime.h>
#include <hip/hip_bf16.h>
#include <math.h>

// ---------------------------------------------------------------- types
typedef __attribute__((ext_vector_type(16))) _Float16 v16h;
typedef __attribute__((ext_vector_type(8)))  _Float16 v8h;
typedef __attribute__((ext_vector_type(8)))  float    v8f;

#define WMMA_F16(A, Bm, C) \
  __builtin_amdgcn_wmma_f32_16x16x32_f16(false, (A), false, (Bm), (short)0, (C), false, false)

// ---------------------------------------------------------------- shapes
static constexpr int   B_    = 4;
static constexpr int   S_    = 1024;
static constexpr int   TD_   = 768;
static constexpr int   VD_   = 1024;
static constexpr int   HD_   = 2048;
static constexpr int   NH_   = 16;
static constexpr int   HDIM_ = 128;
static constexpr int   HD3_  = 3 * HD_;
static constexpr int   BS_   = B_ * S_;
static constexpr float SCALE_  = 0.08838834764831845f;   // 1/sqrt(128)
static constexpr float THRESH_ = 2.0f / 3.0f;

// ---------------------------------------------------------------- helpers
__device__ __forceinline__ v16h ld16(const _Float16* lo, const _Float16* hi) {
  v16h r;
  *reinterpret_cast<v8h*>(&r)       = *reinterpret_cast<const v8h*>(lo);
  *(reinterpret_cast<v8h*>(&r) + 1) = *reinterpret_cast<const v8h*>(hi);
  return r;
}
__device__ __forceinline__ v8f vzero8() {
  v8f z;
#pragma unroll
  for (int i = 0; i < 8; ++i) z[i] = 0.f;
  return z;
}

// Wave computes a 64(M) x 64(N) tile of A[M,K] * Bw[N,K]^T, f16 in / f32 acc.
// Square tile: 16 b128 loads feed 16 WMMAs per 32-K step (~32 FLOP/L2-byte).
__device__ __forceinline__ void wmma_core_64x64(const _Float16* __restrict__ A,
                                                const _Float16* __restrict__ Bw,
                                                int K, int m0, int n0, v8f acc[4][4]) {
  const int lane = threadIdx.x & 31;
  const int l15  = lane & 15;
  const int g    = lane >> 4;
#pragma unroll 1
  for (int kt = 0; kt < K; kt += 32) {
    // speculative prefetch two K-tiles ahead (global_prefetch_b8)
    __builtin_prefetch(A + (size_t)(m0 + lane) * K + kt + 64, 0, 3);
    __builtin_prefetch(Bw + (size_t)(n0 + lane) * K + kt + 64, 0, 3);
    v16h af[4];
#pragma unroll
    for (int mi = 0; mi < 4; ++mi) {
      const _Float16* p = A + (size_t)(m0 + mi * 16 + l15) * K + kt + g * 8;
      af[mi] = ld16(p, p + 16);                 // A-frag: K 0..7 & 16..23 (lanes<16)
    }
    v16h bf[4];
#pragma unroll
    for (int ni = 0; ni < 4; ++ni) {
      const _Float16* p = Bw + (size_t)(n0 + ni * 16 + l15) * K + kt + g * 8;
      bf[ni] = ld16(p, p + 16);                 // B-frag: K 0..7 & 16..23 (lanes<16)
    }
#pragma unroll
    for (int mi = 0; mi < 4; ++mi)
#pragma unroll
      for (int ni = 0; ni < 4; ++ni)
        acc[mi][ni] = WMMA_F16(af[mi], bf[ni], acc[mi][ni]);
  }
}

// ---------------------------------------------------------------- reductions / prep
__global__ void absmean_partial_kernel(const float* __restrict__ w, int count,
                                       float* __restrict__ partial) {
  __shared__ float red[256];
  float s = 0.f;
  for (int i = blockIdx.x * blockDim.x + threadIdx.x; i < count; i += gridDim.x * blockDim.x)
    s += fabsf(w[i]);
  red[threadIdx.x] = s; __syncthreads();
  for (int o = 128; o > 0; o >>= 1) {
    if ((int)threadIdx.x < o) red[threadIdx.x] += red[threadIdx.x + o];
    __syncthreads();
  }
  if (threadIdx.x == 0) partial[blockIdx.x] = red[0];
}

__global__ void absmean_final_kernel(const float* __restrict__ partial, float invCount,
                                     float* __restrict__ scaleOut) {
  __shared__ float red[256];
  red[threadIdx.x] = partial[threadIdx.x]; __syncthreads();
  for (int o = 128; o > 0; o >>= 1) {
    if ((int)threadIdx.x < o) red[threadIdx.x] += red[threadIdx.x + o];
    __syncthreads();
  }
  if (threadIdx.x == 0) {
    float sc = red[0] * invCount;
    scaleOut[0] = fminf(1000.f, fmaxf(1e-5f, sc));
  }
}

__global__ void quant_kernel(const float* __restrict__ w, const float* __restrict__ scalep,
                             _Float16* __restrict__ wq, int count) {
  const float scale = scalep[0], inv = 1.f / scale;
  for (int i = blockIdx.x * blockDim.x + threadIdx.x; i < count; i += gridDim.x * blockDim.x) {
    float wn = fminf(10.f, fmaxf(-10.f, w[i] * inv));
    float q  = (wn > THRESH_) ? 1.f : ((wn < -THRESH_) ? -1.f : 0.f);
    wq[i] = (_Float16)(q * scale);
  }
}

__global__ void cast_f32_to_f16_kernel(const float* __restrict__ in,
                                       _Float16* __restrict__ out, int count) {
  for (int i = blockIdx.x * blockDim.x + threadIdx.x; i < count; i += gridDim.x * blockDim.x)
    out[i] = (_Float16)in[i];
}

// ---------------------------------------------------------------- WMMA GEMMs
__global__ void __launch_bounds__(256)
gemm_bias_f32_kernel(const _Float16* __restrict__ A, const _Float16* __restrict__ Bw,
                     const float* __restrict__ bias, float* __restrict__ out, int N, int K) {
  const int lane = threadIdx.x & 31, l15 = lane & 15, g = lane >> 4;
  const int wid = threadIdx.x >> 5;
  const int m0 = blockIdx.y * 128 + (wid & 1) * 64;
  const int n0 = blockIdx.x * 256 + (wid >> 1) * 64;
  v8f acc[4][4];
#pragma unroll
  for (int mi = 0; mi < 4; ++mi)
#pragma unroll
    for (int ni = 0; ni < 4; ++ni) acc[mi][ni] = vzero8();
  wmma_core_64x64(A, Bw, K, m0, n0, acc);
#pragma unroll
  for (int mi = 0; mi < 4; ++mi)
#pragma unroll
    for (int ni = 0; ni < 4; ++ni)
#pragma unroll
      for (int r = 0; r < 8; ++r) {
        const int row = m0 + mi * 16 + r + 8 * g;
        const int col = n0 + ni * 16 + l15;
        out[(size_t)row * N + col] = acc[mi][ni][r] + bias[col];
      }
}

// QKV GEMM with fused split: Q,K -> [B,NH,S,HDIM] f16 ; V -> transposed [B,NH,HDIM,S] f16
__global__ void __launch_bounds__(256)
gemm_qkv_split_kernel(const _Float16* __restrict__ A, const _Float16* __restrict__ Bw,
                      const float* __restrict__ bias, _Float16* __restrict__ qh,
                      _Float16* __restrict__ kh, _Float16* __restrict__ vt, int K) {
  const int lane = threadIdx.x & 31, l15 = lane & 15, g = lane >> 4;
  const int wid = threadIdx.x >> 5;
  const int m0 = blockIdx.y * 128 + (wid & 1) * 64;
  const int n0 = blockIdx.x * 256 + (wid >> 1) * 64;
  v8f acc[4][4];
#pragma unroll
  for (int mi = 0; mi < 4; ++mi)
#pragma unroll
    for (int ni = 0; ni < 4; ++ni) acc[mi][ni] = vzero8();
  wmma_core_64x64(A, Bw, K, m0, n0, acc);
#pragma unroll
  for (int mi = 0; mi < 4; ++mi)
#pragma unroll
    for (int ni = 0; ni < 4; ++ni)
#pragma unroll
      for (int r = 0; r < 8; ++r) {
        const int row = m0 + mi * 16 + r + 8 * g;   // b*S + s
        const int col = n0 + ni * 16 + l15;         // which*2048 + h*128 + d
        const float v = acc[mi][ni][r] + bias[col];
        const int which = col >> 11;
        const int h = (col >> 7) & 15;
        const int d = col & 127;
        const int b = row >> 10, s = row & 1023;
        const size_t head = (size_t)(b * NH_ + h);
        if (which == 0)      qh[(head * S_ + s) * HDIM_ + d] = (_Float16)v;
        else if (which == 1) kh[(head * S_ + s) * HDIM_ + d] = (_Float16)v;
        else                 vt[(head * HDIM_ + d) * S_ + s] = (_Float16)v;
      }
}

// ---------------------------------------------------------------- flash attention
// One wave per (b, h, 16-row q-tile). Online softmax; QK^T and P*V via WMMA.
__global__ void __launch_bounds__(32)
flash_attn_kernel(const _Float16* __restrict__ qh, const _Float16* __restrict__ kh,
                  const _Float16* __restrict__ vt, const unsigned char* __restrict__ mask,
                  const float* __restrict__ tp, float* __restrict__ ts) {
  __shared__ __align__(16) _Float16 pbuf[16 * 32];
  const int lane = threadIdx.x & 31, l15 = lane & 15, g = lane >> 4;
  const int qtiles = S_ / 16;
  const int qt = blockIdx.x % qtiles;
  const int h  = (blockIdx.x / qtiles) % NH_;
  const int b  = blockIdx.x / (qtiles * NH_);
  const int q0 = qt * 16;
  const size_t head = (size_t)(b * NH_ + h);
  const _Float16* qhp = qh + head * S_ * HDIM_;
  const _Float16* khp = kh + head * S_ * HDIM_;
  const _Float16* vtp = vt + head * HDIM_ * S_;

  v16h qf[4];
#pragma unroll
  for (int c = 0; c < 4; ++c) {
    const _Float16* p = qhp + (size_t)(q0 + l15) * HDIM_ + 32 * c + g * 8;
    qf[c] = ld16(p, p + 16);
  }

  float m8[8], l8[8];
  v8f of[8];
#pragma unroll
  for (int r = 0; r < 8; ++r) { m8[r] = -1e30f; l8[r] = 0.f; }
#pragma unroll
  for (int t = 0; t < 8; ++t) of[t] = vzero8();

#pragma unroll 1
  for (int j = 0; j < S_; j += 32) {
    // ---- scores: two 16x16 tiles, K = HDIM = 4 x 32
    v8f st[2];
#pragma unroll
    for (int n = 0; n < 2; ++n) {
      st[n] = vzero8();
#pragma unroll
      for (int c = 0; c < 4; ++c) {
        const _Float16* p = khp + (size_t)(j + 16 * n + l15) * HDIM_ + 32 * c + g * 8;
        st[n] = WMMA_F16(qf[c], ld16(p, p + 16), st[n]);
      }
    }
    // ---- scale + mask (column fixed per lane)
    const bool keep0 = mask[b * S_ + j + l15] != 0;
    const bool keep1 = mask[b * S_ + j + 16 + l15] != 0;
#pragma unroll
    for (int r = 0; r < 8; ++r) {
      st[0][r] = keep0 ? st[0][r] * SCALE_ : -1e9f;
      st[1][r] = keep1 ? st[1][r] * SCALE_ : -1e9f;
    }
    // ---- row max across the 16-lane group
    float mrow[8];
#pragma unroll
    for (int r = 0; r < 8; ++r) mrow[r] = fmaxf(st[0][r], st[1][r]);
#pragma unroll
    for (int mk = 1; mk <= 8; mk <<= 1)
#pragma unroll
      for (int r = 0; r < 8; ++r) mrow[r] = fmaxf(mrow[r], __shfl_xor(mrow[r], mk, 32));
    float corr[8];
#pragma unroll
    for (int r = 0; r < 8; ++r) {
      const float mn = fmaxf(m8[r], mrow[r]);
      corr[r] = __expf(m8[r] - mn);
      m8[r] = mn;
    }
#pragma unroll
    for (int r = 0; r < 8; ++r) {
      st[0][r] = __expf(st[0][r] - m8[r]);
      st[1][r] = __expf(st[1][r] - m8[r]);
    }
    float rs[8];
#pragma unroll
    for (int r = 0; r < 8; ++r) rs[r] = st[0][r] + st[1][r];
#pragma unroll
    for (int mk = 1; mk <= 8; mk <<= 1)
#pragma unroll
      for (int r = 0; r < 8; ++r) rs[r] += __shfl_xor(rs[r], mk, 32);
#pragma unroll
    for (int r = 0; r < 8; ++r) l8[r] = l8[r] * corr[r] + rs[r];
#pragma unroll
    for (int t = 0; t < 8; ++t)
#pragma unroll
      for (int r = 0; r < 8; ++r) of[t][r] *= corr[r];
    // ---- P: C-layout -> LDS (row-major 16x32) -> A-frag
#pragma unroll
    for (int n = 0; n < 2; ++n)
#pragma unroll
      for (int r = 0; r < 8; ++r)
        pbuf[(r + 8 * g) * 32 + l15 + 16 * n] = (_Float16)st[n][r];
    __syncthreads();
    const v16h pf = ld16(&pbuf[l15 * 32 + g * 8], &pbuf[l15 * 32 + g * 8 + 16]);
    // ---- O += P(16x32) * V(32x128) ; V is pre-transposed so frags are contiguous
#pragma unroll
    for (int t = 0; t < 8; ++t) {
      const _Float16* p = vtp + (size_t)(16 * t + l15) * S_ + j + g * 8;
      of[t] = WMMA_F16(pf, ld16(p, p + 16), of[t]);
    }
    __syncthreads();
  }
  // ---- epilogue: text_self = tp + O / l
#pragma unroll
  for (int t = 0; t < 8; ++t)
#pragma unroll
    for (int r = 0; r < 8; ++r) {
      const int row = q0 + r + 8 * g;
      const size_t idx = (size_t)(b * S_ + row) * HD_ + h * HDIM_ + 16 * t + l15;
      ts[idx] = tp[idx] + of[t][r] / l8[r];
    }
}

// ---------------------------------------------------------------- small VALU kernels
__global__ void ln_kernel(const float* __restrict__ in, const float* __restrict__ aux,
                          int auxShift, const float* __restrict__ gg,
                          const float* __restrict__ bb, float* __restrict__ outF,
                          _Float16* __restrict__ outH) {
  __shared__ float red[256];
  const int row = blockIdx.x;
  const int tid = threadIdx.x;
  float xr[8];
  float s = 0.f, sq = 0.f;
#pragma unroll
  for (int i = 0; i < 8; ++i) {
    const int c = tid + i * 256;
    float x = in[(size_t)row * HD_ + c];
    if (aux) x += aux[(size_t)(row >> auxShift) * HD_ + c];
    xr[i] = x; s += x; sq += x * x;
  }
  red[tid] = s; __syncthreads();
  for (int o = 128; o > 0; o >>= 1) { if (tid < o) red[tid] += red[tid + o]; __syncthreads(); }
  const float mean = red[0] / HD_; __syncthreads();
  red[tid] = sq; __syncthreads();
  for (int o = 128; o > 0; o >>= 1) { if (tid < o) red[tid] += red[tid + o]; __syncthreads(); }
  const float var = red[0] / HD_ - mean * mean;
  const float rstd = rsqrtf(var + 1e-5f);
#pragma unroll
  for (int i = 0; i < 8; ++i) {
    const int c = tid + i * 256;
    const float y = (xr[i] - mean) * rstd * gg[c] + bb[c];
    if (outF) outF[(size_t)row * HD_ + c] = y;
    else      outH[(size_t)row * HD_ + c] = (_Float16)y;
  }
}

// bitnet linear with on-the-fly ternarization (tiny M)
__global__ void tern_proj_kernel(const float* __restrict__ x, const float* __restrict__ w,
                                 const float* __restrict__ bias, const float* __restrict__ scalep,
                                 float* __restrict__ out, int Mrows, int N, int K) {
  const int idx = blockIdx.x * blockDim.x + threadIdx.x;
  if (idx >= Mrows * N) return;
  const int m = idx / N, n = idx % N;
  const float scale = scalep[0], inv = 1.f / scale;
  const float* wr = w + (size_t)n * K;
  const float* xr = x + (size_t)m * K;
  float acc = 0.f;
  for (int k = 0; k < K; ++k) {
    const float wn = fminf(10.f, fmaxf(-10.f, wr[k] * inv));
    const float q  = (wn > THRESH_) ? 1.f : ((wn < -THRESH_) ? -1.f : 0.f);
    acc += xr[k] * q;
  }
  out[idx] = acc * scale + bias[n];
}

// single-query attention per (b,h): i2t_out = softmax(vp . ts^T) @ ts
__global__ void i2t_kernel(const float* __restrict__ vp, const float* __restrict__ ts,
                           const unsigned char* __restrict__ mask, float* __restrict__ out) {
  __shared__ float sc[S_];
  __shared__ float red[256];
  const int bh = blockIdx.x;
  const int b = bh / NH_, h = bh % NH_;
  const int tid = threadIdx.x;
  const float* vph = vp + (size_t)b * HD_ + h * HDIM_;
  const float* tsb = ts + (size_t)b * S_ * HD_ + h * HDIM_;
  float lmax = -1e30f;
  for (int kv = tid; kv < S_; kv += 256) {
    float acc = 0.f;
    const float* r = tsb + (size_t)kv * HD_;
    for (int d = 0; d < HDIM_; ++d) acc += vph[d] * r[d];
    acc *= SCALE_;
    if (!mask[b * S_ + kv]) acc = -1e9f;
    sc[kv] = acc;
    lmax = fmaxf(lmax, acc);
  }
  red[tid] = lmax; __syncthreads();
  for (int o = 128; o > 0; o >>= 1) { if (tid < o) red[tid] = fmaxf(red[tid], red[tid + o]); __syncthreads(); }
  const float smax = red[0]; __syncthreads();
  float lsum = 0.f;
  for (int kv = tid; kv < S_; kv += 256) { const float p = __expf(sc[kv] - smax); sc[kv] = p; lsum += p; }
  red[tid] = lsum; __syncthreads();
  for (int o = 128; o > 0; o >>= 1) { if (tid < o) red[tid] += red[tid + o]; __syncthreads(); }
  const float ssum = red[0]; __syncthreads();
  for (int d = tid; d < HDIM_; d += 256) {
    float acc = 0.f;
    for (int kv = 0; kv < S_; ++kv) acc += sc[kv] * tsb[(size_t)kv * HD_ + d];
    out[(size_t)b * HD_ + h * HDIM_ + d] = acc / ssum;
  }
}

// ---------------------------------------------------------------- launcher
extern "C" void kernel_launch(void* const* d_in, const int* in_sizes, int n_in,
                              void* d_out, int out_size, void* d_ws, size_t ws_size,
                              hipStream_t stream) {
  (void)in_sizes; (void)n_in; (void)out_size; (void)ws_size;
  const float* vision = (const float*)d_in[0];
  const float* textf  = (const float*)d_in[1];
  const unsigned char* mask = (const unsigned char*)d_in[2];
  const float* w_vp  = (const float*)d_in[3];
  const float* b_vp  = (const float*)d_in[4];
  const float* w_tp  = (const float*)d_in[5];
  const float* b_tpv = (const float*)d_in[6];
  const float* w_qkv = (const float*)d_in[7];
  const float* b_qkv = (const float*)d_in[8];
  const float* ln_tg = (const float*)d_in[9];
  const float* ln_tb = (const float*)d_in[10];
  const float* w_out = (const float*)d_in[11];
  const float* b_out = (const float*)d_in[12];
  const float* ln_cg = (const float*)d_in[13];
  const float* ln_cb = (const float*)d_in[14];

  char* ws = (char*)d_ws;
  size_t off = 0;
  auto alloc = [&](size_t bytes) -> void* {
    void* p = ws + off;
    off = (off + bytes + 255) & ~(size_t)255;
    return p;
  };
  float*    scales   = (float*)alloc(4 * sizeof(float));
  float*    partials = (float*)alloc(256 * sizeof(float));
  _Float16* wq_text  = (_Float16*)alloc((size_t)HD_ * TD_ * 2);
  _Float16* wq_qkv   = (_Float16*)alloc((size_t)HD3_ * HD_ * 2);
  _Float16* x_text   = (_Float16*)alloc((size_t)BS_ * TD_ * 2);
  float*    tp       = (float*)alloc((size_t)BS_ * HD_ * 4);
  _Float16* tn       = (_Float16*)alloc((size_t)BS_ * HD_ * 2);
  _Float16* qh       = (_Float16*)alloc((size_t)B_ * NH_ * S_ * HDIM_ * 2);
  _Float16* kh       = (_Float16*)alloc((size_t)B_ * NH_ * S_ * HDIM_ * 2);
  _Float16* vt       = (_Float16*)alloc((size_t)B_ * NH_ * S_ * HDIM_ * 2);
  float*    text_self= (float*)alloc((size_t)BS_ * HD_ * 4);
  float*    vp       = (float*)alloc((size_t)B_ * HD_ * 4);
  float*    i2t_out  = (float*)alloc((size_t)B_ * HD_ * 4);
  float*    proj_i2t = (float*)alloc((size_t)B_ * HD_ * 4);
  float*    proj_t2i = (float*)alloc((size_t)B_ * HD_ * 4);

  // 1) per-tensor abs-mean scales
  absmean_partial_kernel<<<256, 256, 0, stream>>>(w_vp, HD_ * VD_, partials);
  absmean_final_kernel<<<1, 256, 0, stream>>>(partials, 1.f / (HD_ * VD_), scales + 0);
  absmean_partial_kernel<<<256, 256, 0, stream>>>(w_tp, HD_ * TD_, partials);
  absmean_final_kernel<<<1, 256, 0, stream>>>(partials, 1.f / (HD_ * TD_), scales + 1);
  absmean_partial_kernel<<<256, 256, 0, stream>>>(w_qkv, HD3_ * HD_, partials);
  absmean_final_kernel<<<1, 256, 0, stream>>>(partials, 1.f / ((size_t)HD3_ * HD_), scales + 2);
  absmean_partial_kernel<<<256, 256, 0, stream>>>(w_out, HD_ * HD_, partials);
  absmean_final_kernel<<<1, 256, 0, stream>>>(partials, 1.f / (HD_ * HD_), scales + 3);

  // 2) quantize WMMA weights to f16 ternary*scale
  quant_kernel<<<512, 256, 0, stream>>>(w_tp, scales + 1, wq_text, HD_ * TD_);
  quant_kernel<<<2048, 256, 0, stream>>>(w_qkv, scales + 2, wq_qkv, HD3_ * HD_);

  // 3) activations to f16
  cast_f32_to_f16_kernel<<<512, 256, 0, stream>>>(textf, x_text, BS_ * TD_);

  // 4) vision projection (M=4 -> scalar bitnet)
  tern_proj_kernel<<<(B_ * HD_ + 255) / 256, 256, 0, stream>>>(
      vision, w_vp, b_vp, scales + 0, vp, B_, HD_, VD_);

  // 5) tp = text @ Wq_text^T + b   (WMMA, 128x256 block tile)
  gemm_bias_f32_kernel<<<dim3(HD_ / 256, BS_ / 128), 256, 0, stream>>>(
      x_text, wq_text, b_tpv, tp, HD_, TD_);

  // 6) tn = LN(tp) -> f16
  ln_kernel<<<BS_, 256, 0, stream>>>(tp, nullptr, 0, ln_tg, ln_tb, nullptr, tn);

  // 7) qkv GEMM (WMMA) with fused head split + V transpose
  gemm_qkv_split_kernel<<<dim3(HD3_ / 256, BS_ / 128), 256, 0, stream>>>(
      tn, wq_qkv, b_qkv, qh, kh, vt, HD_);

  // 8) flash self-attention -> text_self = tp + attn
  flash_attn_kernel<<<B_ * NH_ * (S_ / 16), 32, 0, stream>>>(qh, kh, vt, mask, tp, text_self);

  // 9) image-to-text single-query attention
  i2t_kernel<<<B_ * NH_, 256, 0, stream>>>(vp, text_self, mask, i2t_out);

  // 10/11) tiny w_out bitnet projections (t2i_out rows are all vp[b])
  tern_proj_kernel<<<(B_ * HD_ + 255) / 256, 256, 0, stream>>>(
      i2t_out, w_out, b_out, scales + 3, proj_i2t, B_, HD_, HD_);
  tern_proj_kernel<<<(B_ * HD_ + 255) / 256, 256, 0, stream>>>(
      vp, w_out, b_out, scales + 3, proj_t2i, B_, HD_, HD_);

  // 12/13) fused LayerNorm epilogues into d_out (fused_vision then fused_text)
  float* outv = (float*)d_out;
  float* outt = outv + (size_t)B_ * HD_;
  ln_kernel<<<B_, 256, 0, stream>>>(vp, proj_i2t, 0, ln_cg, ln_cb, outv, nullptr);
  ln_kernel<<<BS_, 256, 0, stream>>>(text_self, proj_t2i, 10, ln_cg, ln_cb, outt, nullptr);
}
